// STGCN_31482110279805
// MI455X (gfx1250) — matrix operand order repaired and checked by reading, not compiled, not run on hardware
//
#include <hip/hip_runtime.h>
#include <hip/hip_bf16.h>
#include <math.h>

// ---------------- problem constants ----------------
#define HID 512
#define H3  1536   // 3 * HID

typedef __attribute__((ext_vector_type(16))) __bf16 v16bf;
typedef __attribute__((ext_vector_type(8)))  __bf16 v8bf;
typedef __attribute__((ext_vector_type(8)))  float  v8f;

// ---------------- utility kernels ----------------
__global__ void zero_kernel(float* __restrict__ p, long n) {
  long i = (long)blockIdx.x * blockDim.x + threadIdx.x;
  long stride = (long)gridDim.x * blockDim.x;
  for (; i < n; i += stride) p[i] = 0.0f;
}

__global__ void deg_kernel(const int* __restrict__ dst, float* __restrict__ deg, int E) {
  int i = blockIdx.x * blockDim.x + threadIdx.x;
  if (i < E) atomicAdd(&deg[dst[i]], 1.0f);
}

__global__ void rsqrt_kernel(float* __restrict__ inv, int n) {
  int i = blockIdx.x * blockDim.x + threadIdx.x;
  if (i < n) inv[i] = rsqrtf(inv[i] + 1.0f);   // +1 for self loop
}

// ---------------- WMMA GEMM ----------------
// C[M x Ncol] = A[M x K] @ B (+bias, optional relu).
// A fp32 row-major. B: [K x Ncol] (normal) or [Ncol x K] (TRANSB).
// Block = 256 threads = 8 waves; block tile = 16 (M) x 128 (N); wave = one 16x16 tile.
// Per k-step (k=32): stage A tile (16x32) + B slab (32x128) in LDS as bf16 via
// coalesced float4 global loads; lanes read contiguous LDS fragments; then
// v_wmma_f32_16x16x32_bf16. Requires M % 16 == 0, Ncol % 128 == 0, K % 32 == 0.
template <bool TRANSB, bool HAS_BIAS, bool RELU>
__global__ __launch_bounds__(256)
void gemm_wmma(const float* __restrict__ A, const float* __restrict__ B,
               const float* __restrict__ bias, float* __restrict__ C,
               int M, int K, int Ncol) {
  __shared__ __align__(32) __bf16 sA[16 * 32];    // [m][k]        1 KB
  __shared__ __align__(32) __bf16 sB[128 * 32];   // [n_local][k]  8 KB

  const int tid   = threadIdx.x;
  const int wave  = tid >> 5;
  const int lane  = tid & 31;
  const int half  = lane >> 4;      // 0: lanes 0-15, 1: lanes 16-31
  const int l16   = lane & 15;
  const int m0    = blockIdx.x * 16;
  const int nblk0 = blockIdx.y * 128;

  v8f acc = {};
  for (int k0 = 0; k0 < K; k0 += 32) {
    // ---- stage A: 512 floats = 128 float4; threads 0..127 ----
    if (tid < 128) {
      int m  = tid >> 3;            // 0..15
      int k4 = (tid & 7) * 4;       // 0,4,..,28
      const float* pA = A + (long)(m0 + m) * K + k0 + k4;
      __builtin_prefetch(pA + 32, 0, 1);          // next k-step (global_prefetch_b8)
      float4 v = *(const float4*)pA;
      __bf16* d = &sA[m * 32 + k4];
      d[0] = (__bf16)v.x; d[1] = (__bf16)v.y; d[2] = (__bf16)v.z; d[3] = (__bf16)v.w;
    }
    // ---- stage B: 4096 floats = 1024 float4; 256 threads x 4 iters ----
#pragma unroll
    for (int it = 0; it < 4; ++it) {
      int q = it * 256 + tid;       // 0..1023
      if (TRANSB) {
        int nl = q >> 3;            // 0..127
        int k4 = (q & 7) * 4;
        float4 v = *(const float4*)(B + (long)(nblk0 + nl) * K + k0 + k4);
        __bf16* d = &sB[nl * 32 + k4];
        d[0] = (__bf16)v.x; d[1] = (__bf16)v.y; d[2] = (__bf16)v.z; d[3] = (__bf16)v.w;
      } else {
        int k  = q >> 5;            // 0..31
        int n4 = (q & 31) * 4;
        float4 v = *(const float4*)(B + (long)(k0 + k) * Ncol + nblk0 + n4);
        sB[(n4 + 0) * 32 + k] = (__bf16)v.x;
        sB[(n4 + 1) * 32 + k] = (__bf16)v.y;
        sB[(n4 + 2) * 32 + k] = (__bf16)v.z;
        sB[(n4 + 3) * 32 + k] = (__bf16)v.w;
      }
    }
    __syncthreads();

    // ---- fragments (ISA 7.12.2 layouts) ----
    // A: lane m = lane%16; VGPR0-3 -> K = half*8 + 0..7, VGPR4-7 -> +16.
    union { v16bf v; v8bf h[2]; } ua;
    ua.h[0] = *(const v8bf*)&sA[l16 * 32 + half * 8];        // 16B LDS read
    ua.h[1] = *(const v8bf*)&sA[l16 * 32 + 16 + half * 8];   // 16B LDS read
    // B: lane n = wave*16 + lane%16; K = half*8 + 0..7, then +16 (same as A).
    union { v16bf v; v8bf h[2]; } ub;
    ub.h[0] = *(const v8bf*)&sB[(wave * 16 + l16) * 32 + half * 8];        // 16B LDS read
    ub.h[1] = *(const v8bf*)&sB[(wave * 16 + l16) * 32 + 16 + half * 8];   // 16B LDS read
    v16bf bv = ub.v;

    acc = __builtin_amdgcn_wmma_f32_16x16x32_bf16(
        /*neg_a=*/false, ua.v, /*neg_b=*/false, bv,
        /*c_mod=*/(short)0, acc, /*reuse_a=*/false, /*reuse_b=*/false);
    __syncthreads();
  }

  // ---- store: C VGPR r -> M = m0 + half*8 + r; N = nblk0 + wave*16 + lane%16 ----
  const int ncol = nblk0 + wave * 16 + l16;
  const float bval = HAS_BIAS ? bias[ncol] : 0.0f;
  float* __restrict__ Cp = C + (long)(m0 + half * 8) * Ncol + ncol;
#pragma unroll
  for (int r = 0; r < 8; ++r) {
    float v = acc[r] + bval;
    if (RELU) v = fmaxf(v, 0.0f);
    Cp[(long)r * Ncol] = v;
  }
}

// ---------------- edge scatter: agg[dst] += h[src] * inv[src]*inv[dst] ----------------
// 128 threads/edge, float4 gather + 4 global_atomic_add_f32 per thread.
__global__ void scatter_kernel(const int* __restrict__ src, const int* __restrict__ dst,
                               const float* __restrict__ inv, const float* __restrict__ h,
                               float* __restrict__ agg, int E) {
  int e = blockIdx.x;
  if (e >= E) return;
  int s = src[e], d = dst[e];
  float w = inv[s] * inv[d];
  const float* __restrict__ hp = h + (long)s * HID;
  float* __restrict__ ap = agg + (long)d * HID;
  int f = threadIdx.x * 4;                       // blockDim = 128 -> covers 512
  float4 hv = *(const float4*)(hp + f);
  atomicAdd(&ap[f + 0], hv.x * w);
  atomicAdd(&ap[f + 1], hv.y * w);
  atomicAdd(&ap[f + 2], hv.z * w);
  atomicAdd(&ap[f + 3], hv.w * w);
}

// ---------------- self-loop + bias + relu: agg = relu(agg + h*inv^2 + b) ----------------
__global__ void finalize_kernel(float* __restrict__ agg, const float* __restrict__ h,
                                const float* __restrict__ inv, const float* __restrict__ bias,
                                long total) {
  long i = (long)blockIdx.x * blockDim.x + threadIdx.x;
  long stride = (long)gridDim.x * blockDim.x;
  for (; i < total; i += stride) {
    int node = (int)(i >> 9);       // /HID
    int f    = (int)(i & (HID - 1));
    float iv = inv[node];
    float v  = agg[i] + h[i] * iv * iv + bias[f];
    agg[i] = fmaxf(v, 0.0f);
  }
}

// ---------------- GRU persistent scan ----------------
#define GRU_BLOCKS 32
#define IPB 16                  // hidden elements per block (32*16 = 512)
#define TPI 16                  // threads per hidden element
#define KS  (HID / TPI)         // 32 k-elements per thread

__device__ inline void grid_barrier(int* count, int* gen, int nblk) {
  __syncthreads();
  if (threadIdx.x == 0) {
    __threadfence();
    int g = __hip_atomic_load(gen, __ATOMIC_ACQUIRE, __HIP_MEMORY_SCOPE_AGENT);
    int prev = __hip_atomic_fetch_add(count, 1, __ATOMIC_ACQ_REL, __HIP_MEMORY_SCOPE_AGENT);
    if (prev == nblk - 1) {
      __hip_atomic_store(count, 0, __ATOMIC_RELAXED, __HIP_MEMORY_SCOPE_AGENT);
      __hip_atomic_store(gen, g + 1, __ATOMIC_RELEASE, __HIP_MEMORY_SCOPE_AGENT);
    } else {
      while (__hip_atomic_load(gen, __ATOMIC_ACQUIRE, __HIP_MEMORY_SCOPE_AGENT) == g) {
        __builtin_amdgcn_s_sleep(2);
      }
    }
  }
  __syncthreads();
}

__global__ void gru_init(float* __restrict__ hstate, int* __restrict__ sync) {
  int i = threadIdx.x;
  if (i < 2 * HID) hstate[i] = 0.0f;   // zero both ping-pong buffers
  if (i < 2) sync[i] = 0;
}

// 32 persistent blocks; block b owns hidden indices [b*16, b*16+16).
// Its 48 rows of W_hh (gates r,z,n for those indices) live in LDS as bf16 (48 KB).
// Hidden state ping-pongs between hstate[0..511] and hstate[512..1023].
__global__ __launch_bounds__(256)
void gru_scan(const float* __restrict__ prei, const float* __restrict__ Whh,
              const float* __restrict__ bhh, float* __restrict__ ys,
              float* __restrict__ hstate, int* __restrict__ sync, int T) {
  __shared__ __align__(16) __bf16 sW[3 * IPB * HID];   // 48 * 512 bf16 = 48 KB
  __shared__ float sh[HID];

  const int b   = blockIdx.x;
  const int tid = threadIdx.x;
  const int i0  = b * IPB;

  // load this block's W_hh slice into LDS (bf16)
  for (int idx = tid; idx < 3 * IPB * HID; idx += blockDim.x) {
    int r = idx / HID;            // 0..47
    int k = idx - r * HID;
    int g = r / IPB, il = r - g * IPB;
    sW[idx] = (__bf16)Whh[(long)(g * HID + i0 + il) * HID + k];
  }

  const int grp = tid / TPI;      // 0..15  -> local hidden index
  const int sub = tid % TPI;      // 0..15  -> k-slice
  const int i   = i0 + grp;
  const float b_r = bhh[i];
  const float b_z = bhh[HID + i];
  const float b_n = bhh[2 * HID + i];
  const int k0 = sub * KS;
  __syncthreads();

  for (int t = 0; t < T; ++t) {
    const float* __restrict__ hp = hstate + (t & 1) * HID;        // read buffer
    float* __restrict__ hn       = hstate + ((t + 1) & 1) * HID;  // write buffer
    // stage h into LDS
    for (int j = tid; j < HID; j += blockDim.x) sh[j] = hp[j];
    __syncthreads();

    float ar = 0.f, az = 0.f, an = 0.f;
#pragma unroll 8
    for (int kk = 0; kk < KS; ++kk) {
      float hv = sh[k0 + kk];
      ar += (float)sW[(0 * IPB + grp) * HID + k0 + kk] * hv;
      az += (float)sW[(1 * IPB + grp) * HID + k0 + kk] * hv;
      an += (float)sW[(2 * IPB + grp) * HID + k0 + kk] * hv;
    }
#pragma unroll
    for (int off = TPI / 2; off > 0; off >>= 1) {
      ar += __shfl_down(ar, off, TPI);
      az += __shfl_down(az, off, TPI);
      an += __shfl_down(an, off, TPI);
    }
    if (sub == 0) {
      const float* __restrict__ pi = prei + (long)t * H3;
      float r = 1.0f / (1.0f + __expf(-(pi[i] + ar + b_r)));
      float z = 1.0f / (1.0f + __expf(-(pi[HID + i] + az + b_z)));
      float n = tanhf(pi[2 * HID + i] + r * (an + b_n));
      float hnew = (1.0f - z) * n + z * sh[i];
      hn[i] = hnew;
      ys[(long)t * HID + i] = hnew;
    }
    grid_barrier(sync, sync + 1, GRU_BLOCKS);   // publish hn before next read
  }
}

// ---------------- launch ----------------
extern "C" void kernel_launch(void* const* d_in, const int* in_sizes, int n_in,
                              void* d_out, int out_size, void* d_ws, size_t ws_size,
                              hipStream_t stream) {
  const float* x    = (const float*)d_in[0];
  const int*   ei   = (const int*)d_in[1];     // [2, E]: row0 = src, row1 = dst
  const float* W1   = (const float*)d_in[2];
  const float* b1   = (const float*)d_in[3];
  const float* W2   = (const float*)d_in[4];
  const float* b2   = (const float*)d_in[5];
  const float* W_ih = (const float*)d_in[6];   // [1536, 512]
  const float* W_hh = (const float*)d_in[7];   // [1536, 512]
  const float* b_ih = (const float*)d_in[8];
  const float* b_hh = (const float*)d_in[9];
  const float* W_fc = (const float*)d_in[10];
  const float* b_fc = (const float*)d_in[11];
  float* out = (float*)d_out;

  const int Nn = in_sizes[0] / HID;   // 10000 nodes
  const int E  = in_sizes[1] / 2;     // 640000 edges
  const int* src = ei;
  const int* dst = ei + E;

  // workspace layout (floats)
  float* ws     = (float*)d_ws;
  float* hbuf   = ws;                               // N*HID   (x@W result)
  float* aggbuf = hbuf   + (long)Nn * HID;          // N*HID   (layer-1 agg / X1)
  float* hseq   = aggbuf + (long)Nn * HID;          // N*HID   (layer-2 agg / X2, then GRU ys)
  float* prei   = hseq   + (long)Nn * HID;          // N*3H
  float* inv    = prei   + (long)Nn * H3;           // N       (deg -> rsqrt)
  float* hstate = inv    + Nn;                      // 2*HID ping-pong
  int*   syncp  = (int*)(hstate + 2 * HID);         // 2 ints

  // ---- degree normalization ----
  zero_kernel<<<256, 256, 0, stream>>>(inv, Nn);
  deg_kernel<<<(E + 255) / 256, 256, 0, stream>>>(dst, inv, E);
  rsqrt_kernel<<<(Nn + 255) / 256, 256, 0, stream>>>(inv, Nn);

  dim3 g512((Nn + 15) / 16, HID / 128);   // (625, 4)
  dim3 g3h ((Nn + 15) / 16, H3 / 128);    // (625, 12)

  // ---- GCN layer 1 ----
  gemm_wmma<false, false, false><<<g512, 256, 0, stream>>>(x, W1, nullptr, hbuf, Nn, HID, HID);
  zero_kernel<<<2048, 256, 0, stream>>>(aggbuf, (long)Nn * HID);
  scatter_kernel<<<E, 128, 0, stream>>>(src, dst, inv, hbuf, aggbuf, E);
  finalize_kernel<<<4096, 256, 0, stream>>>(aggbuf, hbuf, inv, b1, (long)Nn * HID);

  // ---- GCN layer 2 (agg2 staged in hseq) ----
  gemm_wmma<false, false, false><<<g512, 256, 0, stream>>>(aggbuf, W2, nullptr, hbuf, Nn, HID, HID);
  zero_kernel<<<2048, 256, 0, stream>>>(hseq, (long)Nn * HID);
  scatter_kernel<<<E, 128, 0, stream>>>(src, dst, inv, hbuf, hseq, E);
  finalize_kernel<<<4096, 256, 0, stream>>>(hseq, hbuf, inv, b2, (long)Nn * HID);

  // ---- GRU input projection: prei = X2 @ W_ih^T + b_ih ----
  gemm_wmma<true, true, false><<<g3h, 256, 0, stream>>>(hseq, W_ih, b_ih, prei, Nn, HID, H3);

  // ---- GRU persistent scan (ys overwrites hseq; only prei is read) ----
  gru_init<<<1, 1024, 0, stream>>>(hstate, syncp);
  gru_scan<<<GRU_BLOCKS, 256, 0, stream>>>(prei, W_hh, b_hh, hseq, hstate, syncp, Nn);

  // ---- final FC: out = ys @ W_fc + b_fc ----
  gemm_wmma<false, true, false><<<g512, 256, 0, stream>>>(hseq, W_fc, b_fc, out, Nn, HID, HID);
}
